// DotProductAttentionTransformerMD17_62019327754885
// MI455X (gfx1250) — matrix operand order repaired and checked by reading, not compiled, not run on hardware
//
#include <hip/hip_runtime.h>
#include <math.h>

// ---------------------------------------------------------------------------
// Problem constants (from reference)
// ---------------------------------------------------------------------------
constexpr int NN  = 4096;    // nodes      (multiple of 128)
constexpr int NE  = 98304;   // edges      (multiple of 128)
constexpr int NB  = 128;     // radial basis
constexpr int CS  = 128;     // scalar channels
constexpr int CV  = 64;      // vector channels (x3)
constexpr int CT  = 32;      // tensor channels (x5)
constexpr int NH  = 4;       // heads
constexpr int NL  = 6;       // blocks
constexpr float ATTN_SCALE = 0.09128709291752768f;   // 1/sqrt(32+48+40)
constexpr float INV_SQRT_DEG = 0.25335312f;          // 1/sqrt(15.5793085)
constexpr float RBF_INV_SIGMA = 25.6f;               // NB / CUTOFF

typedef _Float16 h16;
typedef __attribute__((ext_vector_type(16))) _Float16 v16h;
typedef __attribute__((ext_vector_type(8)))  _Float16 v8h;
typedef __attribute__((ext_vector_type(8)))  float    v8f;

static inline int cdiv(int a, int b) { return (a + b - 1) / b; }

// ---------------------------------------------------------------------------
// WMMA GEMM:  C[M,Nc] = epi(A[M,K] @ B[K,Nc] + bias),  K = KT*32 (compile-time)
//   REQUIRES: M multiple of 16 (all M here are multiples of 128) -> no
//   per-element row guards, epilogue is branch-free straight-line code.
//   A: f16 row-major (optionally row-gathered through gidx). The whole A
//      row-fragment (KT k-tiles, <=32 VGPRs/lane) is hoisted into registers
//      ONCE and reused across all N-tiles -> A is read exactly once from HBM.
//   B: pre-packed into per-(ntile,ktile,lane) fragment order (16 halves/lane),
//      streamed in the inner loop (tiny, L2-resident).
//   ACCUM epilogue: 8 old values read into distinct regs (clauseable loads),
//   then add+store -> load latency overlapped instead of 8 serial RMWs.
//   block = 256 threads = 8 waves; wave w owns rows [blk*128 + w*16, +16)
//   Fragment layouts follow CDNA5 ISA (16-bit A 16x32, B 32x16).
// ---------------------------------------------------------------------------
enum { EPI_NONE = 0, EPI_SILU = 1, EPI_SIG = 2 };

template<int KT, int EPI, bool GATHER, bool ACCUM, bool OF32, bool OF16, bool HAS_BIAS>
__global__ __launch_bounds__(256)
void wmma_gemm_kernel(const h16* __restrict__ A, int lda,
                      const int* __restrict__ gidx,
                      const h16* __restrict__ Bp,
                      const float* __restrict__ bias,
                      float* __restrict__ C32, h16* __restrict__ C16, int ldc,
                      int M, int Ncol)
{
  const int lane = threadIdx.x & 31;
  const int wave = threadIdx.x >> 5;
  const int m0   = blockIdx.x * 128 + wave * 16;
  if (m0 >= M) return;                      // wave-uniform: EXEC stays full
  const int nTiles = Ncol >> 4;

  const int row   = m0 + (lane & 15);       // M % 16 == 0 -> always in range
  const int arow  = GATHER ? gidx[row] : row;
  const int khalf = lane >> 4;              // lanes 16-31 hold K+8 / K+24 groups

  union U { v16h v; v8h h[2]; };

  // hoist A fragments: one 16x32 fragment per k-tile, kept in VGPRs
  U afr[KT];
  {
    const h16* ap = A + (size_t)arow * lda + 8 * khalf;
#pragma unroll
    for (int kt = 0; kt < KT; ++kt) {
      afr[kt].h[0] = *(const v8h*)(ap);       // K = kt*32 + 8*khalf + [0..7]
      afr[kt].h[1] = *(const v8h*)(ap + 16);  // K = kt*32 + 16 + 8*khalf + [0..7]
      ap += 32;
    }
  }

  for (int nt = 0; nt < nTiles; ++nt) {
    v8f acc = {};
    const h16* bp = Bp + ((size_t)nt * KT) * 512 + lane * 16;
    U bfr;
#pragma unroll
    for (int kt = 0; kt < KT; ++kt) {
      bfr.h[0] = *(const v8h*)(bp);           // packed fragment: 16 halves/lane
      bfr.h[1] = *(const v8h*)(bp + 8);
      acc = __builtin_amdgcn_wmma_f32_16x16x32_f16(
              false, afr[kt].v, false, bfr.v, (short)0, acc, false, false);
      bp += 512;
    }
    const int col = nt * 16 + (lane & 15);
    const int rb  = 8 * (lane >> 4);          // C/D layout: VGPR r -> M=r (+8)

    float x[8];
#pragma unroll
    for (int r = 0; r < 8; ++r) {
      x[r] = acc[r];
      if (HAS_BIAS) x[r] += bias[col];        // no +0.0f VALU ops when unbiased
      if (EPI == EPI_SILU) x[r] = x[r] / (1.0f + __expf(-x[r]));
      if (EPI == EPI_SIG)  x[r] = 1.0f / (1.0f + __expf(-x[r]));
    }
    const size_t obase = (size_t)(m0 + rb) * ldc + col;
    if (ACCUM) {
      float old[8];
#pragma unroll
      for (int r = 0; r < 8; ++r)             // distinct regs -> clauseable
        old[r] = C32[obase + (size_t)r * ldc];
#pragma unroll
      for (int r = 0; r < 8; ++r)
        C32[obase + (size_t)r * ldc] = old[r] + x[r];
    } else {
#pragma unroll
      for (int r = 0; r < 8; ++r) {
        const size_t o = obase + (size_t)r * ldc;
        if (OF32) C32[o] = x[r];
        if (OF16) C16[o] = (h16)x[r];
      }
    }
  }
}

// Repack f32 weight W[K,Nc] (row-major) into WMMA B-fragment order (f16):
// out[((nt*kTiles+kt)*32 + lane)*16 + j] = W[kt*32 + 8*(lane>>4) + j + (j & 8)][nt*16 + (lane&15)]
__global__ void pack_w_kernel(const float* __restrict__ W, h16* __restrict__ out,
                              int K, int Ncol)
{
  int tid = blockIdx.x * blockDim.x + threadIdx.x;
  int kTiles = K >> 5;
  int total = (Ncol >> 4) * kTiles * 32;
  if (tid >= total) return;
  int lane = tid & 31;
  int kt = (tid >> 5) % kTiles;
  int nt = (tid >> 5) / kTiles;
  int col = nt * 16 + (lane & 15);
  int kb  = kt * 32 + 8 * (lane >> 4);
  h16* dst = out + (size_t)tid * 16;
#pragma unroll
  for (int j = 0; j < 16; ++j)
    dst[j] = (h16)W[(size_t)(kb + j + (j & 8)) * Ncol + col];
}

// ---------------------------------------------------------------------------
// Elementwise / irregular kernels
// ---------------------------------------------------------------------------
__global__ void fill_f32_kernel(float* p, float v, int n) {
  int i = blockIdx.x * blockDim.x + threadIdx.x;
  if (i < n) p[i] = v;
}
__global__ void fill_u32_kernel(unsigned* p, unsigned v, int n) {
  int i = blockIdx.x * blockDim.x + threadIdx.x;
  if (i < n) p[i] = v;
}
__global__ void cvt_f32_h16_kernel(const float* __restrict__ a, h16* __restrict__ b, int n) {
  int i = blockIdx.x * blockDim.x + threadIdx.x;
  if (i < n) b[i] = (h16)a[i];
}

__global__ void geom_kernel(const float* __restrict__ pos,
                            const int* __restrict__ esrc, const int* __restrict__ edst,
                            float* __restrict__ y1, float* __restrict__ y2,
                            float* __restrict__ dist)
{
  int e = blockIdx.x * blockDim.x + threadIdx.x;
  if (e >= NE) return;
  int s = esrc[e], d = edst[e];
  float vx = pos[s*3+0] - pos[d*3+0];
  float vy = pos[s*3+1] - pos[d*3+1];
  float vz = pos[s*3+2] - pos[d*3+2];
  float dd = sqrtf(vx*vx + vy*vy + vz*vz);
  float inv = 1.0f / (dd + 1e-9f);
  float x = vx*inv, y = vy*inv, z = vz*inv;
  const float s3 = 1.7320508075688772f, s15 = 3.872983346207417f;
  const float s5h = 1.118033988749895f, s15h = 1.9364916731037085f;
  y1[e*3+0] = s3*x;  y1[e*3+1] = s3*y;  y1[e*3+2] = s3*z;
  y2[e*5+0] = s15*x*y;
  y2[e*5+1] = s15*y*z;
  y2[e*5+2] = s5h*(3.0f*z*z - 1.0f);
  y2[e*5+3] = s15*x*z;
  y2[e*5+4] = s15h*(x*x - y*y);
  dist[e] = dd;
}

__global__ void rbf_kernel(const float* __restrict__ dist, const float* __restrict__ mu,
                           h16* __restrict__ rbf)
{
  int tid = blockIdx.x * blockDim.x + threadIdx.x;
  if (tid >= NE * NB) return;
  int e = tid >> 7, j = tid & 127;
  float u = (dist[e] - mu[j]) * RBF_INV_SIGMA;
  rbf[tid] = (h16)__expf(-0.5f * u * u);
}

__global__ void embed_kernel(const int* __restrict__ z, const float* __restrict__ emb,
                             float* __restrict__ s)
{
  int tid = blockIdx.x * blockDim.x + threadIdx.x;
  if (tid >= NN * CS) return;
  int n = tid >> 7, c = tid & 127;
  s[tid] = emb[(size_t)z[n] * CS + c];
}

// deg-aggregation scatter: rw[E,224] = [CS | CV | CT] -> s,v,t (atomic)
__global__ void deg_scatter_kernel(const h16* __restrict__ rw,
                                   const float* __restrict__ y1, const float* __restrict__ y2,
                                   const int* __restrict__ edst,
                                   float* __restrict__ s, float* __restrict__ v, float* __restrict__ t)
{
  long tid = (long)blockIdx.x * blockDim.x + threadIdx.x;
  if (tid >= (long)NE * 224) return;
  int e = (int)(tid / 224), c = (int)(tid - (long)e * 224);
  int dn = edst[e];
  float r = INV_SQRT_DEG * (float)rw[(size_t)e * 224 + c];
  if (c < CS) {
    atomicAdd(s + (size_t)dn * CS + c, r);
  } else if (c < CS + CV) {
    int cc = c - CS;
    for (int i = 0; i < 3; ++i)
      atomicAdd(v + (size_t)i * NN * CV + (size_t)dn * CV + cc, r * y1[e*3+i]);
  } else {
    int cc = c - CS - CV;
    for (int j = 0; j < 5; ++j)
      atomicAdd(t + (size_t)j * NN * CT + (size_t)dn * CT + cc, r * y2[e*5+j]);
  }
}

__device__ inline float wred(float x) {
  for (int o = 16; o > 0; o >>= 1) x += __shfl_xor(x, o, 32);
  return x;
}

// Equivariant LayerNorm: one wave32 per node; optionally emits final output.
template<bool FINAL>
__global__ __launch_bounds__(256)
void eqln_kernel(const float* __restrict__ s, const float* __restrict__ v, const float* __restrict__ t,
                 const float* __restrict__ ws, const float* __restrict__ bs,
                 const float* __restrict__ wv, const float* __restrict__ wt,
                 h16* __restrict__ sln, h16* __restrict__ vln, h16* __restrict__ tln,
                 float* __restrict__ outp)
{
  int lane = threadIdx.x & 31, wave = threadIdx.x >> 5;
  int n = blockIdx.x * 8 + wave;
  if (n >= NN) return;
  // scalar channels
  float x[4], sum = 0.f, sq = 0.f;
#pragma unroll
  for (int k = 0; k < 4; ++k) {
    x[k] = s[(size_t)n * CS + lane + 32 * k];
    sum += x[k]; sq += x[k] * x[k];
  }
  sum = wred(sum); sq = wred(sq);
  float mean = sum * (1.0f / CS);
  float var  = fmaxf(sq * (1.0f / CS) - mean * mean, 0.0f);
  float istd = 1.0f / (sqrtf(var) + 1e-5f);
#pragma unroll
  for (int k = 0; k < 4; ++k) {
    int c = lane + 32 * k;
    float o = (x[k] - mean) * istd * ws[c] + bs[c];
    sln[(size_t)n * CS + c] = (h16)o;
    if (FINAL) outp[(size_t)n * 480 + c] = o;
  }
  // vector channels
  float vv[2][3], e2 = 0.f;
#pragma unroll
  for (int k = 0; k < 2; ++k) {
    int c = lane + 32 * k;
#pragma unroll
    for (int i = 0; i < 3; ++i) {
      vv[k][i] = v[(size_t)i * NN * CV + (size_t)n * CV + c];
      e2 += vv[k][i] * vv[k][i];
    }
  }
  e2 = wred(e2);
  float inv_nv = 1.0f / sqrtf(e2 * (1.0f / CV) + 1e-5f);
#pragma unroll
  for (int k = 0; k < 2; ++k) {
    int c = lane + 32 * k;
#pragma unroll
    for (int i = 0; i < 3; ++i) {
      float o = vv[k][i] * inv_nv * wv[c];
      vln[(size_t)i * NN * CV + (size_t)n * CV + c] = (h16)o;
      if (FINAL) outp[(size_t)n * 480 + CS + c * 3 + i] = o;
    }
  }
  // tensor channels
  float tt[5], e2t = 0.f;
  int c = lane;
#pragma unroll
  for (int j = 0; j < 5; ++j) {
    tt[j] = t[(size_t)j * NN * CT + (size_t)n * CT + c];
    e2t += tt[j] * tt[j];
  }
  e2t = wred(e2t);
  float inv_nt = 1.0f / sqrtf(e2t * (1.0f / CT) + 1e-5f);
#pragma unroll
  for (int j = 0; j < 5; ++j) {
    float o = tt[j] * inv_nt * wt[c];
    tln[(size_t)j * NN * CT + (size_t)n * CT + c] = (h16)o;
    if (FINAL) outp[(size_t)n * 480 + CS + CV * 3 + c * 5 + j] = o;
  }
}

// vy[e,c] = sum_i v_ln[src][c][i] * y1[e][i]
__global__ void vy_kernel(const h16* __restrict__ vln, const float* __restrict__ y1,
                          const int* __restrict__ esrc, h16* __restrict__ vy)
{
  int tid = blockIdx.x * blockDim.x + threadIdx.x;
  if (tid >= NE * CV) return;
  int e = tid >> 6, c = tid & 63;
  int sn = esrc[e];
  float acc = 0.f;
#pragma unroll
  for (int i = 0; i < 3; ++i)
    acc += (float)vln[(size_t)i * NN * CV + (size_t)sn * CV + c] * y1[e*3+i];
  vy[tid] = (h16)acc;
}

// messages: rw[E,448] = [r_ss|r_vs|r_vv|r_sv|r_tt|r_st]
__global__ void msg_kernel(const h16* __restrict__ rw,
                           const h16* __restrict__ sln, const h16* __restrict__ vln,
                           const h16* __restrict__ tln,
                           const float* __restrict__ vy2s, const float* __restrict__ svp,
                           const float* __restrict__ stp,
                           const float* __restrict__ y1, const float* __restrict__ y2,
                           const int* __restrict__ esrc,
                           h16* __restrict__ ms, h16* __restrict__ mv, h16* __restrict__ mt)
{
  long tid = (long)blockIdx.x * blockDim.x + threadIdx.x;
  if (tid >= (long)NE * 224) return;
  int e = (int)(tid / 224), c = (int)(tid - (long)e * 224);
  int sn = esrc[e];
  const h16* r = rw + (size_t)e * 448;
  if (c < CS) {
    float o = (float)r[c] * (float)sln[(size_t)sn * CS + c]
            + (float)r[CS + c] * vy2s[(size_t)e * CS + c];
    ms[(size_t)e * CS + c] = (h16)o;
  } else if (c < CS + CV) {
    int cc = c - CS;
    float rvv = (float)r[256 + cc], rsv = (float)r[320 + cc];
    float sv = svp[(size_t)e * CV + cc];
#pragma unroll
    for (int i = 0; i < 3; ++i) {
      float o = rvv * (float)vln[(size_t)i * NN * CV + (size_t)sn * CV + cc]
              + rsv * sv * y1[e*3+i];
      mv[(size_t)i * NE * CV + (size_t)e * CV + cc] = (h16)o;
    }
  } else {
    int cc = c - CS - CV;
    float rtt = (float)r[384 + cc], rst = (float)r[416 + cc];
    float st = stp[(size_t)e * CT + cc];
#pragma unroll
    for (int j = 0; j < 5; ++j) {
      float o = rtt * (float)tln[(size_t)j * NN * CT + (size_t)sn * CT + cc]
              + rst * st * y2[e*5+j];
      mt[(size_t)j * NE * CT + (size_t)e * CT + cc] = (h16)o;
    }
  }
}

__global__ void logits_kernel(const float* __restrict__ qs, const float* __restrict__ qv,
                              const float* __restrict__ qt,
                              const h16* __restrict__ ms, const h16* __restrict__ mv,
                              const h16* __restrict__ mt,
                              const int* __restrict__ edst, float* __restrict__ logits)
{
  int tid = blockIdx.x * blockDim.x + threadIdx.x;
  if (tid >= NE * NH) return;
  int e = tid >> 2, h = tid & 3;
  int dn = edst[e];
  float acc = 0.f;
  {
    const float* q = qs + (size_t)dn * CS + h * 32;
    const h16*   m = ms + (size_t)e  * CS + h * 32;
    for (int c = 0; c < 32; ++c) acc += q[c] * (float)m[c];
  }
  for (int i = 0; i < 3; ++i) {
    const float* q = qv + (size_t)i * NN * CV + (size_t)dn * CV + h * 16;
    const h16*   m = mv + (size_t)i * NE * CV + (size_t)e  * CV + h * 16;
    for (int c = 0; c < 16; ++c) acc += q[c] * (float)m[c];
  }
  for (int j = 0; j < 5; ++j) {
    const float* q = qt + (size_t)j * NN * CT + (size_t)dn * CT + h * 8;
    const h16*   m = mt + (size_t)j * NE * CT + (size_t)e  * CT + h * 8;
    for (int c = 0; c < 8; ++c) acc += q[c] * (float)m[c];
  }
  logits[tid] = acc * ATTN_SCALE;
}

__device__ inline unsigned fenc(float f) {
  unsigned u = __float_as_uint(f);
  return (u & 0x80000000u) ? ~u : (u | 0x80000000u);
}
__device__ inline float fdec(unsigned u) {
  unsigned b = (u & 0x80000000u) ? (u ^ 0x80000000u) : ~u;
  return __uint_as_float(b);
}
__global__ void attn_max_kernel(const float* __restrict__ logits, const int* __restrict__ edst,
                                unsigned* __restrict__ mmax)
{
  int tid = blockIdx.x * blockDim.x + threadIdx.x;
  if (tid >= NE * NH) return;
  int e = tid >> 2, h = tid & 3;
  atomicMax(mmax + (size_t)edst[e] * NH + h, fenc(logits[tid]));
}
__global__ void attn_exp_kernel(float* __restrict__ logits, const int* __restrict__ edst,
                                const unsigned* __restrict__ mmax, float* __restrict__ den)
{
  int tid = blockIdx.x * blockDim.x + threadIdx.x;
  if (tid >= NE * NH) return;
  int e = tid >> 2, h = tid & 3;
  float m = fdec(mmax[(size_t)edst[e] * NH + h]);
  float ev = __expf(logits[tid] - m);
  logits[tid] = ev;
  atomicAdd(den + (size_t)edst[e] * NH + h, ev);
}
__global__ void attn_norm_kernel(const float* __restrict__ enum_, const int* __restrict__ edst,
                                 const float* __restrict__ den, float* __restrict__ alpha)
{
  int tid = blockIdx.x * blockDim.x + threadIdx.x;
  if (tid >= NE * NH) return;
  int e = tid >> 2, h = tid & 3;
  alpha[tid] = enum_[tid] / (den[(size_t)edst[e] * NH + h] + 1e-9f);
}

// alpha-weighted scatter-add of values into node accumulators a = [s|v|t]
__global__ void scatter_kernel(const float* __restrict__ alpha,
                               const h16* __restrict__ vs, const h16* __restrict__ vv,
                               const h16* __restrict__ vt,
                               const int* __restrict__ edst, float* __restrict__ a)
{
  long tid = (long)blockIdx.x * blockDim.x + threadIdx.x;
  if (tid >= (long)NE * 224) return;
  int e = (int)(tid / 224), c = (int)(tid - (long)e * 224);
  int dn = edst[e];
  float* as = a;
  float* av = a + (size_t)NN * CS;
  float* at = a + (size_t)NN * (CS + CV * 3);
  if (c < CS) {
    int h = c >> 5;
    atomicAdd(as + (size_t)dn * CS + c,
              alpha[(size_t)e * NH + h] * (float)vs[(size_t)e * CS + c]);
  } else if (c < CS + CV) {
    int cc = c - CS, h = cc >> 4;
    float al = alpha[(size_t)e * NH + h];
    for (int i = 0; i < 3; ++i)
      atomicAdd(av + (size_t)i * NN * CV + (size_t)dn * CV + cc,
                al * (float)vv[(size_t)i * NE * CV + (size_t)e * CV + cc]);
  } else {
    int cc = c - CS - CV, h = cc >> 3;
    float al = alpha[(size_t)e * NH + h];
    for (int j = 0; j < 5; ++j)
      atomicAdd(at + (size_t)j * NN * CT + (size_t)dn * CT + cc,
                al * (float)vt[(size_t)j * NE * CT + (size_t)e * CT + cc]);
  }
}

// FFN gating: hv = gv * fv (3 planes), ht = gt * ft (5 planes)
__global__ void gate_kernel(const float* __restrict__ gv, const float* __restrict__ fv,
                            const float* __restrict__ gt, const float* __restrict__ ft,
                            h16* __restrict__ hv, h16* __restrict__ ht)
{
  int tid = blockIdx.x * blockDim.x + threadIdx.x;
  const int nv = 3 * NN * CV;
  const int nt = 5 * NN * CT;
  if (tid < nv) {
    int r = tid % (NN * CV);
    hv[tid] = (h16)(gv[r] * fv[tid]);
  } else if (tid < nv + nt) {
    int t2 = tid - nv;
    int r = t2 % (NN * CT);
    ht[t2] = (h16)(gt[r] * ft[t2]);
  }
}

// ---------------------------------------------------------------------------
// Host-side GEMM dispatcher (K -> compile-time KT for register-resident A)
// ---------------------------------------------------------------------------
template<int KT>
static void gemm_kt(hipStream_t st, const h16* A, int lda, const int* gidx,
                    const h16* Bp, const float* bias,
                    float* C32, h16* C16, int ldc, int M, int Nc,
                    int epi, bool accum)
{
  dim3 g(cdiv(M, 128)), b(256);
  if (accum) {   // residual accumulate, never biased
    wmma_gemm_kernel<KT, EPI_NONE, false, true, true, false, false><<<g, b, 0, st>>>(
        A, lda, nullptr, Bp, nullptr, C32, nullptr, ldc, M, Nc);
    return;
  }
  if (gidx) {    // src-gathered edge GEMM, f32 out, never biased
    wmma_gemm_kernel<KT, EPI_NONE, true, false, true, false, false><<<g, b, 0, st>>>(
        A, lda, gidx, Bp, nullptr, C32, nullptr, ldc, M, Nc);
    return;
  }
  if (epi == EPI_SILU) {   // biased silu, f16 out (MLP hidden layers)
    wmma_gemm_kernel<KT, EPI_SILU, false, false, false, true, true><<<g, b, 0, st>>>(
        A, lda, nullptr, Bp, bias, nullptr, C16, ldc, M, Nc);
    return;
  }
  if (epi == EPI_SIG) {    // unbiased sigmoid, f32 out (gates)
    wmma_gemm_kernel<KT, EPI_SIG, false, false, true, false, false><<<g, b, 0, st>>>(
        A, lda, nullptr, Bp, nullptr, C32, nullptr, ldc, M, Nc);
    return;
  }
  if (C16) {
    if (bias)    // radial output layer: biased, f16 out
      wmma_gemm_kernel<KT, EPI_NONE, false, false, false, true, true><<<g, b, 0, st>>>(
          A, lda, nullptr, Bp, bias, nullptr, C16, ldc, M, Nc);
    else         // value projections: unbiased, f16 out
      wmma_gemm_kernel<KT, EPI_NONE, false, false, false, true, false><<<g, b, 0, st>>>(
          A, lda, nullptr, Bp, nullptr, nullptr, C16, ldc, M, Nc);
    return;
  }
  // plain f32 out, unbiased (q projections etc.)
  wmma_gemm_kernel<KT, EPI_NONE, false, false, true, false, false><<<g, b, 0, st>>>(
      A, lda, nullptr, Bp, nullptr, C32, nullptr, ldc, M, Nc);
}

static void gemm(hipStream_t st, const h16* A, int lda, const int* gidx,
                 const h16* Bp, const float* bias,
                 float* C32, h16* C16, int ldc, int M, int Nc, int K,
                 int epi, bool accum)
{
  switch (K >> 5) {
    case 1: gemm_kt<1>(st, A, lda, gidx, Bp, bias, C32, C16, ldc, M, Nc, epi, accum); break;
    case 2: gemm_kt<2>(st, A, lda, gidx, Bp, bias, C32, C16, ldc, M, Nc, epi, accum); break;
    default: gemm_kt<4>(st, A, lda, gidx, Bp, bias, C32, C16, ldc, M, Nc, epi, accum); break;
  }
}

// ---------------------------------------------------------------------------
// kernel_launch
// ---------------------------------------------------------------------------
extern "C" void kernel_launch(void* const* d_in, const int* in_sizes, int n_in,
                              void* d_out, int out_size, void* d_ws, size_t ws_size,
                              hipStream_t stream)
{
  (void)in_sizes; (void)n_in; (void)out_size; (void)ws_size;

  const float* pos  = (const float*)d_in[0];
  const int*   zat  = (const int*)d_in[1];
  // d_in[2] = batch (unused)
  const int*   esrc = (const int*)d_in[3];
  const int*   edst = (const int*)d_in[4];

  // params flattened in jax pytree (sorted-key) order
  int pi = 5;
  auto F = [&](void) { return (const float*)d_in[pi++]; };
  // blocks (each leaf stacked over L)
  const float *bW1s = F(), *bW2s = F(), *bW2t = F(), *bW2v = F();
  const float *bWft = F(), *bWfv = F(), *bWgt = F(), *bWgv = F();
  const float *bWo_s = F(), *bWo_t = F(), *bWo_v = F();
  const float *bWq_s = F(), *bWq_t = F(), *bWq_v = F();
  const float *bWs2t = F(), *bWs2v = F();
  const float *bWval_s = F(), *bWval_t = F(), *bWval_v = F();
  const float *bWv2s = F();
  const float *ln1_bs = F(), *ln1_ws = F(), *ln1_wt = F(), *ln1_wv = F();
  const float *ln2_bs = F(), *ln2_ws = F(), *ln2_wt = F(), *ln2_wv = F();
  const float *radW1 = F(), *radW2 = F(), *radW3 = F();
  const float *radb1 = F(), *radb2 = F(), *radb3 = F();
  // deg_rad
  const float *dW1 = F(), *dW2 = F(), *dW3 = F();
  const float *db1 = F(), *db2 = F(), *db3 = F();
  const float *emb = F();
  const float *lo_bs = F(), *lo_ws = F(), *lo_wt = F(), *lo_wv = F();
  const float *rbf_mu = F();

  // ------------------------- workspace allocator -------------------------
  char* base = (char*)d_ws;
  size_t off = 0;
  auto alloc = [&](size_t bytes) -> void* {
    void* p = base + off;
    off = (off + bytes + 255) & ~(size_t)255;
    return p;
  };

  float* y1    = (float*)alloc(sizeof(float) * NE * 3);
  float* y2    = (float*)alloc(sizeof(float) * NE * 5);
  float* dist  = (float*)alloc(sizeof(float) * NE);
  h16*   rbf16 = (h16*)alloc(sizeof(h16) * (size_t)NE * NB);
  h16*   h1    = (h16*)alloc(sizeof(h16) * (size_t)NE * 64);
  h16*   h2    = (h16*)alloc(sizeof(h16) * (size_t)NE * 64);
  h16*   rw16  = (h16*)alloc(sizeof(h16) * (size_t)NE * 448);
  // node state: s | v(3 planes) | t(5 planes), contiguous
  float* state = (float*)alloc(sizeof(float) * (size_t)NN * 480);
  float* sS = state;
  float* sV = state + (size_t)NN * CS;
  float* sT = state + (size_t)NN * (CS + CV * 3);
  h16* sln = (h16*)alloc(sizeof(h16) * (size_t)NN * CS);
  h16* vln = (h16*)alloc(sizeof(h16) * (size_t)NN * CV * 3);
  h16* tln = (h16*)alloc(sizeof(h16) * (size_t)NN * CT * 5);
  h16*   vy    = (h16*)alloc(sizeof(h16) * (size_t)NE * CV);
  float* vy2s  = (float*)alloc(sizeof(float) * (size_t)NE * CS);
  float* svp   = (float*)alloc(sizeof(float) * (size_t)NE * CV);
  float* stp   = (float*)alloc(sizeof(float) * (size_t)NE * CT);
  h16* ms = (h16*)alloc(sizeof(h16) * (size_t)NE * CS);
  h16* mv = (h16*)alloc(sizeof(h16) * (size_t)NE * CV * 3);
  h16* mt = (h16*)alloc(sizeof(h16) * (size_t)NE * CT * 5);
  float* qs = (float*)alloc(sizeof(float) * (size_t)NN * CS);
  float* qv = (float*)alloc(sizeof(float) * (size_t)NN * CV * 3);
  float* qt = (float*)alloc(sizeof(float) * (size_t)NN * CT * 5);
  float* logits = (float*)alloc(sizeof(float) * (size_t)NE * NH);
  float* alpha  = (float*)alloc(sizeof(float) * (size_t)NE * NH);
  unsigned* mmax = (unsigned*)alloc(sizeof(unsigned) * NN * NH);
  float* den   = (float*)alloc(sizeof(float) * NN * NH);
  h16* vls = (h16*)alloc(sizeof(h16) * (size_t)NE * CS);
  h16* vlv = (h16*)alloc(sizeof(h16) * (size_t)NE * CV * 3);
  h16* vlt = (h16*)alloc(sizeof(h16) * (size_t)NE * CT * 5);
  float* agg   = (float*)alloc(sizeof(float) * (size_t)NN * 480);
  h16*   agg16 = (h16*)alloc(sizeof(h16) * (size_t)NN * 480);
  h16* a16s = agg16;
  h16* a16v = agg16 + (size_t)NN * CS;
  h16* a16t = agg16 + (size_t)NN * (CS + CV * 3);
  float* gvb = (float*)alloc(sizeof(float) * (size_t)NN * CV);
  float* gtb = (float*)alloc(sizeof(float) * (size_t)NN * CT);
  float* fvb = (float*)alloc(sizeof(float) * (size_t)NN * CV * 3);
  float* ftb = (float*)alloc(sizeof(float) * (size_t)NN * CT * 5);
  h16* hs16 = (h16*)alloc(sizeof(h16) * (size_t)NN * CS);
  h16* hv16 = (h16*)alloc(sizeof(h16) * (size_t)NN * CV * 3);
  h16* ht16 = (h16*)alloc(sizeof(h16) * (size_t)NN * CT * 5);

  auto packW = [&](const float* W, int K, int Nc) -> const h16* {
    h16* dst = (h16*)alloc(sizeof(h16) * (size_t)K * Nc);
    int total = (Nc / 16) * (K / 32) * 32;
    pack_w_kernel<<<cdiv(total, 256), 256, 0, stream>>>(W, dst, K, Nc);
    return dst;
  };

  // ------------------------- precompute -------------------------
  geom_kernel<<<cdiv(NE, 256), 256, 0, stream>>>(pos, esrc, edst, y1, y2, dist);
  rbf_kernel<<<cdiv(NE * NB, 256), 256, 0, stream>>>(dist, rbf_mu, rbf16);
  embed_kernel<<<cdiv(NN * CS, 256), 256, 0, stream>>>(zat, emb, sS);
  fill_f32_kernel<<<cdiv(NN * 352, 256), 256, 0, stream>>>(sV, 0.f, NN * 352);

  // deg-radial chain + scatter
  {
    const h16* pW1 = packW(dW1, NB, 64);
    const h16* pW2 = packW(dW2, 64, 64);
    const h16* pW3 = packW(dW3, 64, 224);
    gemm(stream, rbf16, NB, nullptr, pW1, db1, nullptr, h1, 64, NE, 64, NB, EPI_SILU, false);
    gemm(stream, h1, 64, nullptr, pW2, db2, nullptr, h2, 64, NE, 64, 64, EPI_SILU, false);
    gemm(stream, h2, 64, nullptr, pW3, db3, nullptr, rw16, 224, NE, 224, 64, EPI_NONE, false);
    deg_scatter_kernel<<<cdiv(NE * 224, 256), 256, 0, stream>>>(rw16, y1, y2, edst, sS, sV, sT);
  }

  // ------------------------- blocks -------------------------
  for (int l = 0; l < NL; ++l) {
    const h16* pRW1  = packW(radW1 + (size_t)l * NB * 64, NB, 64);
    const h16* pRW2  = packW(radW2 + (size_t)l * 64 * 64, 64, 64);
    const h16* pRW3  = packW(radW3 + (size_t)l * 64 * 448, 64, 448);
    const h16* pWv2s = packW(bWv2s + (size_t)l * CV * CS, CV, CS);
    const h16* pWs2v = packW(bWs2v + (size_t)l * CS * CV, CS, CV);
    const h16* pWs2t = packW(bWs2t + (size_t)l * CS * CT, CS, CT);
    const h16* pWq_s = packW(bWq_s + (size_t)l * CS * CS, CS, CS);
    const h16* pWq_v = packW(bWq_v + (size_t)l * CV * CV, CV, CV);
    const h16* pWq_t = packW(bWq_t + (size_t)l * CT * CT, CT, CT);
    const h16* pWvs  = packW(bWval_s + (size_t)l * CS * CS, CS, CS);
    const h16* pWvv  = packW(bWval_v + (size_t)l * CV * CV, CV, CV);
    const h16* pWvt  = packW(bWval_t + (size_t)l * CT * CT, CT, CT);
    const h16* pWo_s = packW(bWo_s + (size_t)l * CS * CS, CS, CS);
    const h16* pWo_v = packW(bWo_v + (size_t)l * CV * CV, CV, CV);
    const h16* pWo_t = packW(bWo_t + (size_t)l * CT * CT, CT, CT);
    const h16* pW1s  = packW(bW1s + (size_t)l * CS * CS, CS, CS);
    const h16* pWgv  = packW(bWgv + (size_t)l * CS * CV, CS, CV);
    const h16* pWgt  = packW(bWgt + (size_t)l * CS * CT, CS, CT);
    const h16* pWfv  = packW(bWfv + (size_t)l * CV * CV, CV, CV);
    const h16* pWft  = packW(bWft + (size_t)l * CT * CT, CT, CT);
    const h16* pW2s  = packW(bW2s + (size_t)l * CS * CS, CS, CS);
    const h16* pW2v  = packW(bW2v + (size_t)l * CV * CV, CV, CV);
    const h16* pW2t  = packW(bW2t + (size_t)l * CT * CT, CT, CT);

    // ln1
    eqln_kernel<false><<<NN / 8, 256, 0, stream>>>(
        sS, sV, sT, ln1_ws + (size_t)l * CS, ln1_bs + (size_t)l * CS,
        ln1_wv + (size_t)l * CV, ln1_wt + (size_t)l * CT, sln, vln, tln, nullptr);

    // radial MLP -> rw[E,448]
    gemm(stream, rbf16, NB, nullptr, pRW1, radb1 + (size_t)l * 64, nullptr, h1, 64, NE, 64, NB, EPI_SILU, false);
    gemm(stream, h1, 64, nullptr, pRW2, radb2 + (size_t)l * 64, nullptr, h2, 64, NE, 64, 64, EPI_SILU, false);
    gemm(stream, h2, 64, nullptr, pRW3, radb3 + (size_t)l * 448, nullptr, rw16, 448, NE, 448, 64, EPI_NONE, false);

    // message pieces
    vy_kernel<<<cdiv(NE * CV, 256), 256, 0, stream>>>(vln, y1, esrc, vy);
    gemm(stream, vy, CV, nullptr, pWv2s, nullptr, vy2s, nullptr, CS, NE, CS, CV, EPI_NONE, false);
    gemm(stream, sln, CS, esrc, pWs2v, nullptr, svp, nullptr, CV, NE, CV, CS, EPI_NONE, false);
    gemm(stream, sln, CS, esrc, pWs2t, nullptr, stp, nullptr, CT, NE, CT, CS, EPI_NONE, false);
    msg_kernel<<<cdiv(NE * 224, 256), 256, 0, stream>>>(
        rw16, sln, vln, tln, vy2s, svp, stp, y1, y2, esrc, ms, mv, mt);

    // queries
    gemm(stream, sln, CS, nullptr, pWq_s, nullptr, qs, nullptr, CS, NN, CS, CS, EPI_NONE, false);
    for (int i = 0; i < 3; ++i)
      gemm(stream, vln + (size_t)i * NN * CV, CV, nullptr, pWq_v, nullptr,
           qv + (size_t)i * NN * CV, nullptr, CV, NN, CV, CV, EPI_NONE, false);
    for (int j = 0; j < 5; ++j)
      gemm(stream, tln + (size_t)j * NN * CT, CT, nullptr, pWq_t, nullptr,
           qt + (size_t)j * NN * CT, nullptr, CT, NN, CT, CT, EPI_NONE, false);

    // attention logits + segment softmax over dst
    logits_kernel<<<cdiv(NE * NH, 256), 256, 0, stream>>>(qs, qv, qt, ms, mv, mt, edst, logits);
    fill_u32_kernel<<<cdiv(NN * NH, 256), 256, 0, stream>>>(mmax, 0u, NN * NH);
    fill_f32_kernel<<<cdiv(NN * NH, 256), 256, 0, stream>>>(den, 0.f, NN * NH);
    attn_max_kernel<<<cdiv(NE * NH, 256), 256, 0, stream>>>(logits, edst, mmax);
    attn_exp_kernel<<<cdiv(NE * NH, 256), 256, 0, stream>>>(logits, edst, mmax, den);
    attn_norm_kernel<<<cdiv(NE * NH, 256), 256, 0, stream>>>(logits, edst, den, alpha);

    // value projections
    gemm(stream, ms, CS, nullptr, pWvs, nullptr, nullptr, vls, CS, NE, CS, CS, EPI_NONE, false);
    for (int i = 0; i < 3; ++i)
      gemm(stream, mv + (size_t)i * NE * CV, CV, nullptr, pWvv, nullptr, nullptr,
           vlv + (size_t)i * NE * CV, CV, NE, CV, CV, EPI_NONE, false);
    for (int j = 0; j < 5; ++j)
      gemm(stream, mt + (size_t)j * NE * CT, CT, nullptr, pWvt, nullptr, nullptr,
           vlt + (size_t)j * NE * CT, CT, NE, CT, CT, EPI_NONE, false);

    // aggregate + output projections (residual accumulate)
    fill_f32_kernel<<<cdiv(NN * 480, 256), 256, 0, stream>>>(agg, 0.f, NN * 480);
    scatter_kernel<<<cdiv(NE * 224, 256), 256, 0, stream>>>(alpha, vls, vlv, vlt, edst, agg);
    cvt_f32_h16_kernel<<<cdiv(NN * 480, 256), 256, 0, stream>>>(agg, agg16, NN * 480);
    gemm(stream, a16s, CS, nullptr, pWo_s, nullptr, sS, nullptr, CS, NN, CS, CS, EPI_NONE, true);
    for (int i = 0; i < 3; ++i)
      gemm(stream, a16v + (size_t)i * NN * CV, CV, nullptr, pWo_v, nullptr,
           sV + (size_t)i * NN * CV, nullptr, CV, NN, CV, CV, EPI_NONE, true);
    for (int j = 0; j < 5; ++j)
      gemm(stream, a16t + (size_t)j * NN * CT, CT, nullptr, pWo_t, nullptr,
           sT + (size_t)j * NN * CT, nullptr, CT, NN, CT, CT, EPI_NONE, true);

    // ln2 + gated FFN (residual accumulate)
    eqln_kernel<false><<<NN / 8, 256, 0, stream>>>(
        sS, sV, sT, ln2_ws + (size_t)l * CS, ln2_bs + (size_t)l * CS,
        ln2_wv + (size_t)l * CV, ln2_wt + (size_t)l * CT, sln, vln, tln, nullptr);
    gemm(stream, sln, CS, nullptr, pW1s, nullptr, nullptr, hs16, CS, NN, CS, CS, EPI_SILU, false);
    gemm(stream, sln, CS, nullptr, pWgv, nullptr, gvb, nullptr, CV, NN, CV, CS, EPI_SIG, false);
    gemm(stream, sln, CS, nullptr, pWgt, nullptr, gtb, nullptr, CT, NN, CT, CS, EPI_SIG, false);
    for (int i = 0; i < 3; ++i)
      gemm(stream, vln + (size_t)i * NN * CV, CV, nullptr, pWfv, nullptr,
           fvb + (size_t)i * NN * CV, nullptr, CV, NN, CV, CV, EPI_NONE, false);
    for (int j = 0; j < 5; ++j)
      gemm(stream, tln + (size_t)j * NN * CT, CT, nullptr, pWft, nullptr,
           ftb + (size_t)j * NN * CT, nullptr, CT, NN, CT, CT, EPI_NONE, false);
    gate_kernel<<<cdiv(3 * NN * CV + 5 * NN * CT, 256), 256, 0, stream>>>(gvb, fvb, gtb, ftb, hv16, ht16);
    gemm(stream, hs16, CS, nullptr, pW2s, nullptr, sS, nullptr, CS, NN, CS, CS, EPI_NONE, true);
    for (int i = 0; i < 3; ++i)
      gemm(stream, hv16 + (size_t)i * NN * CV, CV, nullptr, pW2v, nullptr,
           sV + (size_t)i * NN * CV, nullptr, CV, NN, CV, CV, EPI_NONE, true);
    for (int j = 0; j < 5; ++j)
      gemm(stream, ht16 + (size_t)j * NN * CT, CT, nullptr, pW2t, nullptr,
           sT + (size_t)j * NN * CT, nullptr, CT, NN, CT, CT, EPI_NONE, true);
  }

  // ------------------------- final eqLN + concat output -------------------------
  eqln_kernel<true><<<NN / 8, 256, 0, stream>>>(
      sS, sV, sT, lo_ws, lo_bs, lo_wv, lo_wt, sln, vln, tln, (float*)d_out);
}
